// FastAttention_83571473645963
// MI455X (gfx1250) — matrix operand order repaired and checked by reading, not compiled, not run on hardware
//
#include <hip/hip_runtime.h>
#include <hip/hip_bf16.h>

// ---------------------------------------------------------------------------
// Problem constants
// ---------------------------------------------------------------------------
#define BB 2
#define LL 2048
#define DMODEL 512
#define HH 8
#define DK 64
#define RANK 32
#define RFF 64
#define KMAX 64
#define NROW (BB*LL)          // 4096
#define RFFSCALE 0.1767766953f // sqrt(2/64)

typedef _Float16 v8h  __attribute__((ext_vector_type(8)));
typedef _Float16 v16h __attribute__((ext_vector_type(16)));
typedef float    v8f  __attribute__((ext_vector_type(8)));

__device__ __forceinline__ v16h cat16(v8h a, v8h b) {
    return __builtin_shufflevector(a, b, 0,1,2,3,4,5,6,7,8,9,10,11,12,13,14,15);
}

// ---------------------------------------------------------------------------
// Split-precision f16 WMMA GEMM:  C[M,N] = A[M,K] @ B[K,N] (+ bias epilogue)
// A,B fp32 row-major; staged to LDS as hi/lo f16 pairs (Markidis split) so the
// product is fp32-accurate while running on v_wmma_f32_16x16x32_f16.
// Block = 256 threads = 8 waves; tile 128(M) x 64(N), K-step 32.
// mode 0: C = acc + bias[n]
// mode 1: C = cos(acc + bias[n]) * sqrt(2/RFF)        (RFF feature epilogue)
// ---------------------------------------------------------------------------
__global__ __launch_bounds__(256)
void wmma_gemm_split(const float* __restrict__ A, int lda,
                     const float* __restrict__ B, int ldb,
                     const float* __restrict__ bias,
                     float* __restrict__ C, int ldc,
                     int K, int mode)
{
    __shared__ __align__(16) _Float16 As_hi[128*32];
    __shared__ __align__(16) _Float16 As_lo[128*32];
    __shared__ __align__(16) _Float16 Bs_hi[64*32];   // transposed: [n][k]
    __shared__ __align__(16) _Float16 Bs_lo[64*32];

    const int tid  = threadIdx.x;
    const int wave = tid >> 5;
    const int lane = tid & 31;
    const int m0 = blockIdx.x * 128;
    const int n0 = blockIdx.y * 64;

    v8f acc[4] = {};

    const int arow = tid >> 1;          // 0..127
    const int acol = (tid & 1) * 16;    // 0 or 16
    const int brow = tid >> 3;          // 0..31
    const int bcol = (tid & 7) * 8;     // 0..56

    const int ar = lane & 15;
    const int hi = lane >> 4;           // 0 or 1

    for (int kk = 0; kk < K; kk += 32) {
        // ---- stage A tile (128x32), split into hi/lo halves ----
        {
            const float* ap = A + (size_t)(m0 + arow) * lda + kk + acol;
            if (kk + 32 < K) __builtin_prefetch(ap + 32, 0, 0);
            #pragma unroll
            for (int j = 0; j < 4; ++j) {
                float4 f = *(const float4*)(ap + j*4);
                int o = arow*32 + acol + j*4;
                _Float16 h0=(_Float16)f.x, h1=(_Float16)f.y,
                         h2=(_Float16)f.z, h3=(_Float16)f.w;
                As_hi[o+0]=h0; As_hi[o+1]=h1; As_hi[o+2]=h2; As_hi[o+3]=h3;
                As_lo[o+0]=(_Float16)(f.x-(float)h0);
                As_lo[o+1]=(_Float16)(f.y-(float)h1);
                As_lo[o+2]=(_Float16)(f.z-(float)h2);
                As_lo[o+3]=(_Float16)(f.w-(float)h3);
            }
        }
        // ---- stage B tile (32x64) transposed to [n][k] ----
        {
            const float* bp = B + (size_t)(kk + brow) * ldb + n0 + bcol;
            if (kk + 32 < K) __builtin_prefetch(bp + 32*ldb, 0, 0);
            #pragma unroll
            for (int j = 0; j < 2; ++j) {
                float4 f = *(const float4*)(bp + j*4);
                int nb = bcol + j*4;
                _Float16 h0=(_Float16)f.x, h1=(_Float16)f.y,
                         h2=(_Float16)f.z, h3=(_Float16)f.w;
                Bs_hi[(nb+0)*32 + brow]=h0;
                Bs_hi[(nb+1)*32 + brow]=h1;
                Bs_hi[(nb+2)*32 + brow]=h2;
                Bs_hi[(nb+3)*32 + brow]=h3;
                Bs_lo[(nb+0)*32 + brow]=(_Float16)(f.x-(float)h0);
                Bs_lo[(nb+1)*32 + brow]=(_Float16)(f.y-(float)h1);
                Bs_lo[(nb+2)*32 + brow]=(_Float16)(f.z-(float)h2);
                Bs_lo[(nb+3)*32 + brow]=(_Float16)(f.w-(float)h3);
            }
        }
        __syncthreads();

        // ---- A fragments (16x32 f16 layout: lanes 0-15 K{0..7,16..23},
        //      lanes 16-31 K{8..15,24..31}; row = lane&15) ----
        const int aoff = (wave*16 + ar)*32 + hi*8;
        v16h ahi = cat16(*(const v8h*)(As_hi + aoff), *(const v8h*)(As_hi + aoff + 16));
        v16h alo = cat16(*(const v8h*)(As_lo + aoff), *(const v8h*)(As_lo + aoff + 16));

        #pragma unroll
        for (int nt = 0; nt < 4; ++nt) {
            // B fragment: N = lane&15, K = {hi*8 + i, 16 + hi*8 + i} (same order as A)
            const int boff = (nt*16 + ar)*32 + hi*8;
            v16h bhi = cat16(*(const v8h*)(Bs_hi + boff), *(const v8h*)(Bs_hi + boff + 16));
            v16h blo = cat16(*(const v8h*)(Bs_lo + boff), *(const v8h*)(Bs_lo + boff + 16));
            acc[nt] = __builtin_amdgcn_wmma_f32_16x16x32_f16(
                          false, ahi, false, bhi, (short)0, acc[nt], false, false);
            acc[nt] = __builtin_amdgcn_wmma_f32_16x16x32_f16(
                          false, alo, false, bhi, (short)0, acc[nt], false, false);
            acc[nt] = __builtin_amdgcn_wmma_f32_16x16x32_f16(
                          false, ahi, false, blo, (short)0, acc[nt], false, false);
        }
        __syncthreads();
    }

    // ---- epilogue: C layout M = r + 8*hi, N = lane&15 ----
    #pragma unroll
    for (int nt = 0; nt < 4; ++nt) {
        const int col = n0 + nt*16 + ar;
        const float bv = bias[col];
        #pragma unroll
        for (int r = 0; r < 8; ++r) {
            const int row = m0 + wave*16 + r + hi*8;
            float s = acc[nt][r] + bv;
            float v = (mode == 1) ? cosf(s) * RFFSCALE : s;
            C[(size_t)row*ldc + col] = v;
        }
    }
}

// ---------------------------------------------------------------------------
// LSH hash: pack 4 bucket ids (5 bits each) per (side, head, row)
// ---------------------------------------------------------------------------
__global__ __launch_bounds__(256)
void hash_kernel(const float* __restrict__ qproj, const float* __restrict__ kproj,
                 const float* __restrict__ lsh,
                 unsigned* __restrict__ qhash, unsigned* __restrict__ khash)
{
    int id = blockIdx.x * blockDim.x + threadIdx.x;   // 2*H*NROW = 65536
    if (id >= 2*HH*NROW) return;
    int side = id >> 15;
    int rem  = id & 32767;
    int h  = rem >> 12;
    int bl = rem & (NROW-1);

    const float* x  = (side ? kproj : qproj) + (size_t)bl*DMODEL + h*DK;
    const float* rv = lsh + (size_t)h*DK*4;

    float s0=0.f, s1=0.f, s2=0.f, s3=0.f;
    #pragma unroll 4
    for (int d = 0; d < DK; ++d) {
        if (x[d] > 0.f) {
            float4 r = *(const float4*)(rv + d*4);
            s0 += r.x; s1 += r.y; s2 += r.z; s3 += r.w;
        }
    }
    unsigned b0 = ((int)floorf(s0 * 0.125f)) & 31;
    unsigned b1 = ((int)floorf(s1 * 0.125f)) & 31;
    unsigned b2 = ((int)floorf(s2 * 0.125f)) & 31;
    unsigned b3 = ((int)floorf(s3 * 0.125f)) & 31;
    unsigned pk = b0 | (b1<<5) | (b2<<10) | (b3<<15);
    (side ? khash : qhash)[h*NROW + bl] = pk;
}

// ---------------------------------------------------------------------------
// Candidate search: one wave32 per query; ballot + popc prefix compaction of
// the first KMAX ascending hash-matching key indices.
// ---------------------------------------------------------------------------
__global__ __launch_bounds__(256)
void cand_kernel(const unsigned* __restrict__ qhash,
                 const unsigned* __restrict__ khash,
                 int* __restrict__ cands, int* __restrict__ counts)
{
    int wid  = blockIdx.x * (blockDim.x >> 5) + (threadIdx.x >> 5); // 0..32767
    int lane = threadIdx.x & 31;
    int h  = wid >> 12;
    int bl = wid & (NROW-1);
    int b  = bl >> 11;

    unsigned qh = qhash[h*NROW + bl];
    const unsigned* kh = khash + h*NROW + b*LL;
    int* outc = cands + (size_t)wid * KMAX;

    int cnt = 0;
    for (int j0 = 0; j0 < LL && cnt < KMAX; j0 += 32) {
        bool pred = (kh[j0 + lane] == qh);
        unsigned mask = __builtin_amdgcn_ballot_w32(pred);
        int pos = __popc(mask & ((1u << lane) - 1u));
        if (pred) {
            int slot = cnt + pos;
            if (slot < KMAX) outc[slot] = j0 + lane;
        }
        cnt += __popc(mask);
    }
    if (lane == 0) counts[wid] = (cnt < KMAX) ? cnt : KMAX;
}

// ---------------------------------------------------------------------------
// Sparse attention: one 64-thread block per (head, query).
// sim -> softmax -> ctx over <=64 candidates; writes ctx into the head-major
// concat buffer that feeds the fused (U*V*Wo) output GEMM.
// ---------------------------------------------------------------------------
__global__ __launch_bounds__(64)
void attn_kernel(const float* __restrict__ qfeat, const float* __restrict__ kfeat,
                 const float* __restrict__ vproj,
                 const int* __restrict__ cands, const int* __restrict__ counts,
                 float* __restrict__ ctx)
{
    __shared__ int   cand_s[KMAX];
    __shared__ float attn_s[KMAX];
    __shared__ float qf_s[RFF];

    const int qid = blockIdx.x;          // 0..32767
    const int h   = qid >> 12;
    const int bl  = qid & (NROW-1);
    const int b   = bl >> 11;
    const int t   = threadIdx.x;

    const int cnt = counts[qid];
    cand_s[t] = (t < cnt) ? cands[(size_t)qid*KMAX + t] : 0;
    qf_s[t]   = qfeat[((size_t)h*NROW + bl)*RFF + t];
    __syncthreads();

    if (t < cnt) {
        const float* kf = kfeat + ((size_t)h*NROW + b*LL + cand_s[t]) * RFF;
        float s = 0.f;
        #pragma unroll 8
        for (int f = 0; f < RFF; ++f) s += qf_s[f] * kf[f];
        attn_s[t] = s * 0.125f;          // / sqrt(RFF)
    }
    __syncthreads();

    if (t == 0) {                        // 64-wide serial softmax (tiny)
        float m = -3.0e38f;
        for (int c = 0; c < cnt; ++c) m = fmaxf(m, attn_s[c]);
        float sum = 0.f;
        for (int c = 0; c < cnt; ++c) { float e = expf(attn_s[c] - m); attn_s[c] = e; sum += e; }
        float inv = (cnt > 0) ? 1.f / sum : 0.f;
        for (int c = 0; c < cnt; ++c) attn_s[c] *= inv;
        for (int c = cnt; c < KMAX; ++c) attn_s[c] = 0.f;
    }
    __syncthreads();

    float acc = 0.f;
    const float* vb = vproj + (size_t)(b*LL)*DMODEL + h*DK + t;
    for (int c = 0; c < cnt; ++c)
        acc += attn_s[c] * vb[(size_t)cand_s[c]*DMODEL];
    ctx[(size_t)bl*DMODEL + h*DK + t] = acc;
}

// ---------------------------------------------------------------------------
// Fold the output path:  T[h] = V[h] @ Wo[h*512:(h+1)*512, :]   [32,512]
//                        M[h] = U[h] @ T[h]                      [64,512]
// so   out = concat_ctx[4096,512] @ M[512,512] + bo   (single WMMA GEMM)
// ---------------------------------------------------------------------------
__global__ __launch_bounds__(256)
void fold_T_kernel(const float* __restrict__ V, const float* __restrict__ Wo,
                   float* __restrict__ T)
{
    int id = blockIdx.x * blockDim.x + threadIdx.x;   // H*32*512 = 131072
    if (id >= HH*RANK*DMODEL) return;
    int h = id / (RANK*DMODEL);
    int r = (id / DMODEL) % RANK;
    int n = id % DMODEL;
    const float* vr = V + ((size_t)h*RANK + r)*DMODEL;
    const float* wc = Wo + (size_t)h*DMODEL*DMODEL + n;
    float s = 0.f;
    #pragma unroll 4
    for (int d = 0; d < DMODEL; ++d) s += vr[d] * wc[(size_t)d*DMODEL];
    T[id] = s;
}

__global__ __launch_bounds__(256)
void fold_M_kernel(const float* __restrict__ U, const float* __restrict__ T,
                   float* __restrict__ M)
{
    int id = blockIdx.x * blockDim.x + threadIdx.x;   // H*64*512 = 262144
    if (id >= HH*DK*DMODEL) return;
    int h = id / (DK*DMODEL);
    int x = (id / DMODEL) % DK;
    int n = id % DMODEL;
    const float* ur = U + ((size_t)h*DK + x)*RANK;
    const float* tc = T + (size_t)h*RANK*DMODEL + n;
    float s = 0.f;
    #pragma unroll
    for (int r = 0; r < RANK; ++r) s += ur[r] * tc[(size_t)r*DMODEL];
    M[((size_t)h*DK + x)*DMODEL + n] = s;   // row h*64+x of the fused matrix
}

// ---------------------------------------------------------------------------
extern "C" void kernel_launch(void* const* d_in, const int* in_sizes, int n_in,
                              void* d_out, int out_size, void* d_ws, size_t ws_size,
                              hipStream_t stream)
{
    const float* query = (const float*)d_in[0];
    const float* key   = (const float*)d_in[1];
    const float* value = (const float*)d_in[2];
    const float* Wq    = (const float*)d_in[3];
    const float* bq    = (const float*)d_in[4];
    const float* Wk    = (const float*)d_in[5];
    const float* bk    = (const float*)d_in[6];
    const float* Wv    = (const float*)d_in[7];
    const float* bv    = (const float*)d_in[8];
    const float* U     = (const float*)d_in[9];
    const float* V     = (const float*)d_in[10];
    const float* omega = (const float*)d_in[11];
    const float* rffb  = (const float*)d_in[12];
    const float* lsh   = (const float*)d_in[13];
    const float* Wo    = (const float*)d_in[14];
    const float* bo    = (const float*)d_in[15];
    float* out = (float*)d_out;

    // ---- workspace carve (all 16B aligned) ----
    float* ws    = (float*)d_ws;
    float* qproj = ws;                                  // [4096,512]
    float* kproj = qproj + (size_t)NROW*DMODEL;
    float* vproj = kproj + (size_t)NROW*DMODEL;
    float* qfeat = vproj + (size_t)NROW*DMODEL;         // [H,4096,64]
    float* kfeat = qfeat + (size_t)HH*NROW*RFF;
    float* ctx   = kfeat + (size_t)HH*NROW*RFF;         // [4096,512] head-major cols
    float* Tbuf  = ctx   + (size_t)NROW*DMODEL;         // [H,32,512]
    float* Mbuf  = Tbuf  + (size_t)HH*RANK*DMODEL;      // [512,512]
    unsigned* qhash = (unsigned*)(Mbuf + (size_t)DMODEL*DMODEL);
    unsigned* khash = qhash + HH*NROW;
    int* counts = (int*)(khash + HH*NROW);
    int* cands  = counts + HH*NROW;                     // [32768,64]

    dim3 blk(256);

    // 1) QKV projections (split-f16 WMMA, fp32-accurate)
    dim3 gProj(NROW/128, DMODEL/64);
    wmma_gemm_split<<<gProj, blk, 0, stream>>>(query, DMODEL, Wq, DMODEL, bq, qproj, DMODEL, DMODEL, 0);
    wmma_gemm_split<<<gProj, blk, 0, stream>>>(key,   DMODEL, Wk, DMODEL, bk, kproj, DMODEL, DMODEL, 0);
    wmma_gemm_split<<<gProj, blk, 0, stream>>>(value, DMODEL, Wv, DMODEL, bv, vproj, DMODEL, DMODEL, 0);

    // 2) RFF features for ALL queries and ALL keys (per head) via WMMA + cos epilogue
    dim3 gFeat(NROW/128, 1);
    for (int h = 0; h < HH; ++h) {
        wmma_gemm_split<<<gFeat, blk, 0, stream>>>(qproj + h*DK, DMODEL,
                                                   omega + (size_t)h*DK*RFF, RFF,
                                                   rffb + h*RFF,
                                                   qfeat + (size_t)h*NROW*RFF, RFF, DK, 1);
        wmma_gemm_split<<<gFeat, blk, 0, stream>>>(kproj + h*DK, DMODEL,
                                                   omega + (size_t)h*DK*RFF, RFF,
                                                   rffb + h*RFF,
                                                   kfeat + (size_t)h*NROW*RFF, RFF, DK, 1);
    }

    // 3) LSH hashes (packed 4x5 bits)
    hash_kernel<<<(2*HH*NROW + 255)/256, blk, 0, stream>>>(qproj, kproj, lsh, qhash, khash);

    // 4) candidate search: one wave per (head, query)
    cand_kernel<<<(HH*NROW)/8, blk, 0, stream>>>(qhash, khash, cands, counts);

    // 5) sparse attention -> ctx (head-major concat layout)
    attn_kernel<<<HH*NROW, dim3(64), 0, stream>>>(qfeat, kfeat, vproj, cands, counts, ctx);

    // 6) fold U,V,Wo into one [512,512] matrix
    fold_T_kernel<<<(HH*RANK*DMODEL + 255)/256, blk, 0, stream>>>(V, Wo, Tbuf);
    fold_M_kernel<<<(HH*DK*DMODEL  + 255)/256, blk, 0, stream>>>(U, Tbuf, Mbuf);

    // 7) fused output GEMM: out = ctx @ M + bo   (WMMA)
    wmma_gemm_split<<<gProj, blk, 0, stream>>>(ctx, DMODEL, Mbuf, DMODEL, bo, out, DMODEL, DMODEL, 0);
}
